// Grouped_Query_Attention_38199439130910
// MI455X (gfx1250) — matrix operand order repaired and checked by reading, not compiled, not run on hardware
//
#include <hip/hip_runtime.h>
#include <hip/hip_bf16.h>

// ---------------------------------------------------------------------------
// GQA block for MI455X (gfx1250, wave32, WMMA).
// Shapes: B=4, T=2048, E=1024, Hq=16, Hkv=4, D=64, F=1536 (q 1024 | k 256 | v 256)
// ---------------------------------------------------------------------------

typedef _Float16 f16;
typedef __attribute__((ext_vector_type(16))) _Float16 v16h;
typedef __attribute__((ext_vector_type(8)))  _Float16 v8h;
typedef __attribute__((ext_vector_type(8)))  float    v8f;

#define WMMA_F16(a, b, c) \
  __builtin_amdgcn_wmma_f32_16x16x32_f16(false, (a), false, (b), (short)0, (c), false, false)

union AFrag { v16h v; v8h h[2]; };

// A-matrix fragment (16x32 f16, M x K, K contiguous in memory):
// lane L: m = L&15, kb = (L>>4)*8 ; halves 0-7 = K kb..kb+7, halves 8-15 = K kb+16..kb+23
__device__ __forceinline__ v16h load_a_frag(const f16* __restrict__ base, int ld, int lane) {
  const int m  = lane & 15;
  const int kb = (lane >> 4) * 8;
  AFrag a;
  a.h[0] = *(const v8h*)(base + (size_t)m * ld + kb);
  a.h[1] = *(const v8h*)(base + (size_t)m * ld + kb + 16);
  return a.v;
}

// B-matrix fragment (32x16 f16, K x N) sourced from W stored as W[n][k] (K contiguous):
// lane L: n = L&15, kb = (L>>4)*8 ; halves 0-7 = K kb..kb+7, halves 8-15 = K kb+16..kb+23 (W[n][k])
__device__ __forceinline__ v16h load_b_frag(const f16* __restrict__ base, int ld, int lane) {
  const int n  = lane & 15;
  const int kb = (lane >> 4) * 8;
  AFrag b;
  b.h[0] = *(const v8h*)(base + (size_t)n * ld + kb);
  b.h[1] = *(const v8h*)(base + (size_t)n * ld + kb + 16);
  return b.v;
}

// ---------------------------------------------------------------------------
// DPP 16-lane row reductions (lanes 0-15 and 16-31 reduce independently —
// exactly the WMMA C-layout row groups). Pure VALU, no LDS permute.
// ---------------------------------------------------------------------------
#if __has_builtin(__builtin_amdgcn_update_dpp)
template <int CTRL>
__device__ __forceinline__ float dpp_mov(float v) {
  return __int_as_float(
      __builtin_amdgcn_update_dpp(0, __float_as_int(v), CTRL, 0xf, 0xf, true));
}
__device__ __forceinline__ float row16_max(float v) {
  v = fmaxf(v, dpp_mov<0xB1>(v));    // quad_perm [1,0,3,2]  : xor 1
  v = fmaxf(v, dpp_mov<0x4E>(v));    // quad_perm [2,3,0,1]  : xor 2
  v = fmaxf(v, dpp_mov<0x141>(v));   // row_half_mirror      : xor 7
  v = fmaxf(v, dpp_mov<0x140>(v));   // row_mirror           : xor 15
  return v;
}
__device__ __forceinline__ float row16_sum(float v) {
  v += dpp_mov<0xB1>(v);
  v += dpp_mov<0x4E>(v);
  v += dpp_mov<0x141>(v);
  v += dpp_mov<0x140>(v);
  return v;
}
#else
__device__ __forceinline__ float row16_max(float v) {
  for (int off = 1; off < 16; off <<= 1) v = fmaxf(v, __shfl_xor(v, off, 32));
  return v;
}
__device__ __forceinline__ float row16_sum(float v) {
  for (int off = 1; off < 16; off <<= 1) v += __shfl_xor(v, off, 32);
  return v;
}
#endif

// ---------------------------------------------------------------------------
// fp32 -> fp16 convert
// ---------------------------------------------------------------------------
__global__ void f32_to_f16_kernel(const float* __restrict__ in, f16* __restrict__ out, int n) {
  for (int i = blockIdx.x * blockDim.x + threadIdx.x; i < n; i += gridDim.x * blockDim.x)
    out[i] = (f16)in[i];
}

// ---------------------------------------------------------------------------
// Generic WMMA GEMM: Out[m,n] = sum_k A[m,k] * W[n,k]
// One wave computes a 32x64 C macro-tile (2 M-tiles x 4 N-tiles, 8 WMMA / K-step).
// ---------------------------------------------------------------------------
template <bool OUT_F32>
__global__ __launch_bounds__(256) void gemm_wmma_kernel(
    const f16* __restrict__ A, const f16* __restrict__ W, void* __restrict__ Out,
    int M, int N, int K) {
  const int lane   = threadIdx.x & 31;
  const int ntiles = N >> 6;
  const int wave   = blockIdx.x * 8 + (threadIdx.x >> 5);
  if (wave >= (M >> 5) * ntiles) return;
  const int tm = wave / ntiles;
  const int tn = wave % ntiles;
  const int m0 = tm << 5;
  const int n0 = tn << 6;

  v8f acc[2][4] = {};
  for (int kc = 0; kc < K; kc += 32) {
    const v16h a0 = load_a_frag(A + (size_t)m0 * K + kc,        K, lane);
    const v16h a1 = load_a_frag(A + (size_t)(m0 + 16) * K + kc, K, lane);
#pragma unroll
    for (int nt = 0; nt < 4; ++nt) {
      const v16h b = load_b_frag(W + (size_t)(n0 + nt * 16) * K + kc, K, lane);
      acc[0][nt] = WMMA_F16(a0, b, acc[0][nt]);
      acc[1][nt] = WMMA_F16(a1, b, acc[1][nt]);
    }
  }

  const int n  = lane & 15;
  const int mb = (lane >> 4) * 8;
#pragma unroll
  for (int mt = 0; mt < 2; ++mt)
#pragma unroll
    for (int nt = 0; nt < 4; ++nt)
#pragma unroll
      for (int j = 0; j < 8; ++j) {
        const size_t idx = (size_t)(m0 + mt * 16 + mb + j) * N + (size_t)(n0 + nt * 16 + n);
        if (OUT_F32) ((float*)Out)[idx] = acc[mt][nt][j];
        else         ((f16*)Out)[idx]   = (f16)acc[mt][nt][j];
      }
}

// ---------------------------------------------------------------------------
// RoPE + layout shuffle.
//  qkv_f16 : [B*T, 1536] rows (q 0..1023 | k 1024..1279 | v 1280..1535)
//  q_out   : [B][16][T][64]  (rope'd, pre-scaled by log2(e)/sqrt(64) for
//                             exp2-domain softmax)
//  k_out   : [B][ 4][T][64]  (rope'd)
//  vT_out  : [B][ 4][64][T]  (transposed so PV B-fragments are contiguous)
// ---------------------------------------------------------------------------
__global__ __launch_bounds__(256) void rope_split_kernel(
    const f16* __restrict__ qkv, f16* __restrict__ qo,
    f16* __restrict__ ko, f16* __restrict__ vT) {
  const int bt = blockIdx.x;           // B*T
  const int b  = bt >> 11;
  const int t  = bt & 2047;
  const f16* __restrict__ row = qkv + (size_t)bt * 1536;
  const float QSCALE = 0.125f * 1.44269504088896340736f;   // 1/sqrt(D) * log2(e)

  // q heads: 16 heads * 32 rotation pairs
  for (int i = threadIdx.x; i < 512; i += 256) {
    const int h = i >> 5, d = i & 31;
    const float ang = (float)t * __powf(10000.0f, -(float)d * (1.0f / 32.0f));
    float c, s; __sincosf(ang, &s, &c);
    const float t1 = (float)row[h * 64 + d];
    const float t2 = (float)row[h * 64 + d + 32];
    const size_t base = (((size_t)(b * 16 + h)) * 2048 + t) * 64;
    qo[base + d]      = (f16)((t1 * c - t2 * s) * QSCALE);
    qo[base + d + 32] = (f16)((t2 * c + t1 * s) * QSCALE);
  }
  // k heads: 4 heads * 32 pairs
  for (int i = threadIdx.x; i < 128; i += 256) {
    const int h = i >> 5, d = i & 31;
    const float ang = (float)t * __powf(10000.0f, -(float)d * (1.0f / 32.0f));
    float c, s; __sincosf(ang, &s, &c);
    const float t1 = (float)row[1024 + h * 64 + d];
    const float t2 = (float)row[1024 + h * 64 + d + 32];
    const size_t base = (((size_t)(b * 4 + h)) * 2048 + t) * 64;
    ko[base + d]      = (f16)(t1 * c - t2 * s);
    ko[base + d + 32] = (f16)(t2 * c + t1 * s);
  }
  // v: 4 heads * 64 dims, store transposed [b][h][d][t]
  for (int i = threadIdx.x; i < 256; i += 256) {
    const int h = i >> 6, d = i & 63;
    vT[(((size_t)(b * 4 + h)) * 64 + d) * 2048 + t] = row[1280 + h * 64 + d];
  }
}

// ---------------------------------------------------------------------------
// One flash-attention chunk of NH*16 key columns starting at s0, for a
// 32-query-row wave (2 M-tiles share every K/V fragment -> 2x arithmetic
// intensity on K/V traffic). Scores are in log2 domain (exp2 softmax).
// MASKED=false is the steady state (no cmp/cndmask at all).
// ---------------------------------------------------------------------------
template <int NH, bool MASKED>
__device__ __forceinline__ void attn_chunk(
    int s0, int q0, int lane, int n, int mb,
    const f16* __restrict__ kbh, const f16* __restrict__ vbh, f16* __restrict__ pbuf,
    const v16h (&qa)[2][2],
    v8f (&o)[2][4], float (&mrow)[2][8], float (&lrow)[2][8]) {
  // ---- scores: 2 x NH 16x16 tiles over D=64; K fragments reused across M ----
  v8f sc[2][NH];
#pragma unroll
  for (int hf = 0; hf < NH; ++hf) {
    const f16* krow = kbh + (size_t)(s0 + hf * 16) * 64;
    const v16h kb0 = load_b_frag(krow,      64, lane);
    const v16h kb1 = load_b_frag(krow + 32, 64, lane);
#pragma unroll
    for (int mt = 0; mt < 2; ++mt) {
      v8f cc = {};
      cc = WMMA_F16(qa[mt][0], kb0, cc);
      cc = WMMA_F16(qa[mt][1], kb1, cc);
      sc[mt][hf] = cc;
    }
  }

  // ---- online softmax (lane owns rows mt*16+mb+j, column n of each tile) ----
#pragma unroll
  for (int mt = 0; mt < 2; ++mt) {
#pragma unroll
    for (int j = 0; j < 8; ++j) {
      if (MASKED) {
        const int row = q0 + mt * 16 + mb + j;
#pragma unroll
        for (int hf = 0; hf < NH; ++hf)
          if (s0 + hf * 16 + n > row) sc[mt][hf][j] = -1e30f;
      }
      float mx = sc[mt][0][j];
#pragma unroll
      for (int hf = 1; hf < NH; ++hf) mx = fmaxf(mx, sc[mt][hf][j]);
      mx = row16_max(mx);
      const float mnew = fmaxf(mrow[mt][j], mx);

      float p[NH], rs = 0.0f;
#pragma unroll
      for (int hf = 0; hf < NH; ++hf) { p[hf] = exp2f(sc[mt][hf][j] - mnew); rs += p[hf]; }
      rs = row16_sum(rs);

      const float alpha = exp2f(mrow[mt][j] - mnew);
      lrow[mt][j] = lrow[mt][j] * alpha + rs;
      mrow[mt][j] = mnew;
#pragma unroll
      for (int nt = 0; nt < 4; ++nt) o[mt][nt][j] *= alpha;

      // stage P (C layout -> LDS row-major [32][NH*16])
#pragma unroll
      for (int hf = 0; hf < NH; ++hf)
        pbuf[(mt * 16 + mb + j) * (NH * 16) + hf * 16 + n] = (f16)p[hf];
    }
  }

  asm volatile("s_wait_dscnt 0" ::: "memory");  // LDS stores visible before re-read

  // ---- PV: P as A-fragments (16x32), V^T rows as contiguous B-fragments ----
#pragma unroll
  for (int kk = 0; kk < NH / 2; ++kk) {
    AFrag pa[2];
#pragma unroll
    for (int mt = 0; mt < 2; ++mt) {
      const f16* pr = pbuf + (mt * 16 + (lane & 15)) * (NH * 16) + kk * 32 + (lane >> 4) * 8;
      pa[mt].h[0] = *(const v8h*)(pr);
      pa[mt].h[1] = *(const v8h*)(pr + 16);
    }
#pragma unroll
    for (int nt = 0; nt < 4; ++nt) {
      const v16h vb = load_b_frag(vbh + (size_t)(nt * 16) * 2048 + s0 + kk * 32, 2048, lane);
      o[0][nt] = WMMA_F16(pa[0].v, vb, o[0][nt]);
      o[1][nt] = WMMA_F16(pa[1].v, vb, o[1][nt]);
    }
  }
}

// ---------------------------------------------------------------------------
// Flash attention (causal, GQA). One wave per (b, q-head, 32-query block).
// All 8 waves of a block share the same q-block -> uniform trip count.
// Steady state: unmasked 64-column chunks (32 WMMA each); only the 1-2
// diagonal 32-column chunks carry the causal cmp/cndmask cost.
// attn_out: [B*T, 1024] f16 rows (head-major within row), K-contiguous for GEMM3.
// ---------------------------------------------------------------------------
__global__ __launch_bounds__(256) void attn_flash_kernel(
    const f16* __restrict__ q, const f16* __restrict__ k,
    const f16* __restrict__ vT, f16* __restrict__ attn) {
  __shared__ f16 pstage[8][32 * 64];     // 4KB per wave

  const int lane = threadIdx.x & 31;
  const int w    = threadIdx.x >> 5;
  const int q0   = blockIdx.x * 32;              // query block start (uniform in block)
  const int hb   = blockIdx.y * 8 + w;           // 0..63 == b*16 + qh
  const int b    = hb >> 4;
  const int qh   = hb & 15;
  const int kvh  = qh >> 2;

  const f16* __restrict__ qbase = q  + (((size_t)hb) * 2048 + q0) * 64;
  const f16* __restrict__ kbh   = k  + ((size_t)(b * 4 + kvh)) * 2048 * 64;
  const f16* __restrict__ vbh   = vT + ((size_t)(b * 4 + kvh)) * 64 * 2048;
  f16* __restrict__ pbuf = &pstage[w][0];

  v16h qa[2][2];
#pragma unroll
  for (int mt = 0; mt < 2; ++mt) {
    qa[mt][0] = load_a_frag(qbase + (size_t)mt * 16 * 64,      64, lane);  // D 0..31
    qa[mt][1] = load_a_frag(qbase + (size_t)mt * 16 * 64 + 32, 64, lane);  // D 32..63
  }

  const int n  = lane & 15;
  const int mb = (lane >> 4) * 8;

  v8f o[2][4] = {};
  float mrow[2][8], lrow[2][8];
#pragma unroll
  for (int mt = 0; mt < 2; ++mt)
#pragma unroll
    for (int j = 0; j < 8; ++j) { mrow[mt][j] = -1e30f; lrow[mt][j] = 0.0f; }

  int s0 = 0;
  const int nfull64 = q0 >> 6;
  for (int c = 0; c < nfull64; ++c, s0 += 64)       // fully below diagonal, 64-wide
    attn_chunk<4, false>(s0, q0, lane, n, mb, kbh, vbh, pbuf, qa, o, mrow, lrow);
  for (; s0 + 32 <= q0; s0 += 32)                    // fully below diagonal, 32-wide
    attn_chunk<2, false>(s0, q0, lane, n, mb, kbh, vbh, pbuf, qa, o, mrow, lrow);
  for (; s0 < q0 + 32; s0 += 32)                     // diagonal chunks (masked)
    attn_chunk<2, true>(s0, q0, lane, n, mb, kbh, vbh, pbuf, qa, o, mrow, lrow);

  // ---- epilogue: normalize, scatter into [B*T, 1024] f16 ----
#pragma unroll
  for (int mt = 0; mt < 2; ++mt)
#pragma unroll
    for (int j = 0; j < 8; ++j) {
      const float inv = 1.0f / lrow[mt][j];
      const int row = q0 + mt * 16 + mb + j;
      f16* orow = attn + ((size_t)(b * 2048 + row)) * 1024 + qh * 64 + n;
#pragma unroll
      for (int nt = 0; nt < 4; ++nt) orow[nt * 16] = (f16)(o[mt][nt][j] * inv);
    }
}

// ---------------------------------------------------------------------------
// Launch
// ---------------------------------------------------------------------------
extern "C" void kernel_launch(void* const* d_in, const int* in_sizes, int n_in,
                              void* d_out, int out_size, void* d_ws, size_t ws_size,
                              hipStream_t stream) {
  (void)in_sizes; (void)n_in; (void)out_size; (void)ws_size;
  constexpr int B = 4, T = 2048, E = 1024, F = 1536, Hq = 16, Hkv = 4, D = 64;
  constexpr int BT = B * T;                         // 8192

  const float* x    = (const float*)d_in[0];
  const float* wqkv = (const float*)d_in[1];
  const float* wout = (const float*)d_in[2];
  float* out = (float*)d_out;

  // workspace layout (all 256B-aligned)
  char* ws = (char*)d_ws;
  constexpr size_t OFF_XH    = 0;                                   // 16 MB
  constexpr size_t OFF_WQKVH = OFF_XH    + (size_t)BT * E * 2;      //  3 MB
  constexpr size_t OFF_WOUTH = OFF_WQKVH + (size_t)F  * E * 2;      //  2 MB
  constexpr size_t OFF_QKVH  = OFF_WOUTH + (size_t)E  * E * 2;      // 24 MB
  constexpr size_t OFF_Q     = OFF_QKVH  + (size_t)BT * F * 2;      // 16 MB
  constexpr size_t OFF_K     = OFF_Q     + (size_t)B * Hq  * T * D * 2; // 4 MB
  constexpr size_t OFF_VT    = OFF_K     + (size_t)B * Hkv * T * D * 2; // 4 MB
  constexpr size_t OFF_ATTN  = OFF_VT    + (size_t)B * Hkv * T * D * 2; // 16 MB

  f16* xh    = (f16*)(ws + OFF_XH);
  f16* wqkvh = (f16*)(ws + OFF_WQKVH);
  f16* wouth = (f16*)(ws + OFF_WOUTH);
  f16* qkvh  = (f16*)(ws + OFF_QKVH);
  f16* qf    = (f16*)(ws + OFF_Q);
  f16* kf    = (f16*)(ws + OFF_K);
  f16* vTf   = (f16*)(ws + OFF_VT);
  f16* attnh = (f16*)(ws + OFF_ATTN);

  // 1) fp32 -> fp16
  f32_to_f16_kernel<<<2048, 256, 0, stream>>>(x,    xh,    BT * E);
  f32_to_f16_kernel<<<512,  256, 0, stream>>>(wqkv, wqkvh, F * E);
  f32_to_f16_kernel<<<512,  256, 0, stream>>>(wout, wouth, E * E);

  // 2) QKV projection: [8192,1024] x [1536,1024]^T -> f16 [8192,1536]
  {
    const int waves = (BT >> 5) * (F >> 6);         // 6144
    gemm_wmma_kernel<false><<<waves / 8, 256, 0, stream>>>(xh, wqkvh, qkvh, BT, F, E);
  }

  // 3) RoPE + split + V transpose
  rope_split_kernel<<<BT, 256, 0, stream>>>(qkvh, qf, kf, vTf);

  // 4) Causal GQA flash attention -> f16 [8192,1024]
  {
    dim3 grid(T / 32, (B * Hq) / 8);                // 64 x 8 blocks, 8 waves each
    attn_flash_kernel<<<grid, 256, 0, stream>>>(qf, kf, vTf, attnh);
  }

  // 5) Output projection: [8192,1024] x [1024,1024]^T -> f32 d_out
  {
    const int waves = (BT >> 5) * (E >> 6);         // 4096
    gemm_wmma_kernel<true><<<waves / 8, 256, 0, stream>>>(attnh, wouth, out, BT, E, E);
  }
}
